// AttentionHeads_13005160972651
// MI455X (gfx1250) — matrix operand order repaired and checked by reading, not compiled, not run on hardware
//
#include <hip/hip_runtime.h>

typedef __attribute__((ext_vector_type(16))) _Float16 v16h;
typedef __attribute__((ext_vector_type(8)))  float    v8f;

#define N_EDGES_C 150000
#define N_FEAT    64
#define N_HEADS   8
#define N_RBF_C   20
#define CUTOFF_F  5.0f
#define PI_F      3.14159265358979323846f

// Edges per workgroup (4 subtiles of 16)
#define E_TILE 64

// LDS row strides (halves). 72*2=144B and 40*2=80B: 16B-aligned rows whose
// start banks are distinct multiples of 4 -> conflict-free b128 access.
#define XSTRIDE 72
#define ESTRIDE 40

// f16 weight workspace layout (element counts)
#define WQ16_N (N_HEADS * N_FEAT * N_FEAT)   // 32768
#define WK16_N (N_HEADS * N_FEAT * N_FEAT)   // 32768
#define WD16_N (N_HEADS * N_FEAT * 32)       // 16384 (K padded 20 -> 32)
#define WS_NEED ((size_t)(WQ16_N + WK16_N + WD16_N) * sizeof(_Float16))

union FragU { v16h v; uint4 q[2]; };
union Pack4 { uint2 u; _Float16 h[4]; };
union Pack8 { uint4 u; _Float16 h[8]; };

__device__ __forceinline__ float silu_f(float x) {
    return x / (1.0f + __expf(-x));
}

// Load two runs of 8 halves at p and p+16 (16B-aligned) as one fragment: 2 x b128.
__device__ __forceinline__ v16h load_frag16(const _Float16* p) {
    FragU f;
    f.q[0] = *(const uint4*)(p);
    f.q[1] = *(const uint4*)(p + 16);
    return f.v;
}

// ---- One-shot weight pre-conversion: f32 -> f16 (Wdk zero-padded K 20->32) ----
__global__ __launch_bounds__(256) void convert_weights(
    const float* __restrict__ W_q, const float* __restrict__ W_k,
    const float* __restrict__ W_dk,
    _Float16* __restrict__ wq16, _Float16* __restrict__ wk16,
    _Float16* __restrict__ wd16)
{
    int i = blockIdx.x * 256 + threadIdx.x;      // 0 .. 32767
    if (i < WQ16_N) {
        wq16[i] = (_Float16)W_q[i];
        wk16[i] = (_Float16)W_k[i];
    }
    if (i < WD16_N) {
        int k = i & 31;
        wd16[i] = (k < N_RBF_C) ? (_Float16)W_dk[(i >> 5) * N_RBF_C + k]
                                : (_Float16)0.0f;
    }
}

template <bool USE_WS>
__global__ __launch_bounds__(256) void painn_attn_wmma(
    const float* __restrict__ dist,
    const int*   __restrict__ nbrs,
    const float* __restrict__ x_i,
    const float* __restrict__ W_q,
    const float* __restrict__ b_q,
    const float* __restrict__ W_k,
    const float* __restrict__ b_k,
    const float* __restrict__ W_dk,
    const float* __restrict__ b_dk,
    const _Float16* __restrict__ wq16,
    const _Float16* __restrict__ wk16,
    const _Float16* __restrict__ wd16,
    float*       __restrict__ out)
{
    __shared__ _Float16 sXi[E_TILE][XSTRIDE];
    __shared__ _Float16 sXj[E_TILE][XSTRIDE];
    __shared__ _Float16 sEF[E_TILE][ESTRIDE];   // K padded 20 -> 32 with zeros

    const int tid   = threadIdx.x;
    const int ebase = blockIdx.x * E_TILE;

    // ---- Stage gathered features: 64 edges x 16 float4-chunks = 1024 tasks ----
    #pragma unroll
    for (int it = 0; it < 4; ++it) {
        int task = tid + 256 * it;          // 0..1023
        int el   = task >> 4;               // local edge 0..63
        int ch   = task & 15;               // float4 chunk 0..15
        int e    = ebase + el;
        int ri = 0, rj = 0;
        if (e < N_EDGES_C) { ri = nbrs[2 * e]; rj = nbrs[2 * e + 1]; }
        float4 vi = *(const float4*)(x_i + ri * N_FEAT + ch * 4);
        float4 vj = *(const float4*)(x_i + rj * N_FEAT + ch * 4);
        Pack4 pi, pj;
        pi.h[0] = (_Float16)vi.x; pi.h[1] = (_Float16)vi.y;
        pi.h[2] = (_Float16)vi.z; pi.h[3] = (_Float16)vi.w;
        pj.h[0] = (_Float16)vj.x; pj.h[1] = (_Float16)vj.y;
        pj.h[2] = (_Float16)vj.z; pj.h[3] = (_Float16)vj.w;
        *(uint2*)&sXi[el][ch * 4] = pi.u;   // 8B-aligned b64 store
        *(uint2*)&sXj[el][ch * 4] = pj.u;
    }

    // ---- Stage RBF * cosine envelope: 64 edges x 4 groups of 8 slots ----
    {
        int el = tid >> 2;
        int s0 = (tid & 3) * 8;
        int e  = ebase + el;
        float d   = (e < N_EDGES_C) ? dist[e] : 1.0f;
        float env = (d < CUTOFF_F) ? 0.5f * (__cosf(PI_F * d / CUTOFF_F) + 1.0f) : 0.0f;
        float inv = env / d;
        Pack8 p;
        #pragma unroll
        for (int j = 0; j < 8; ++j) {
            int r = s0 + j;
            float val = (r < N_RBF_C)
                      ? __sinf((float)(r + 1) * PI_F * d / CUTOFF_F) * inv
                      : 0.0f;
            p.h[j] = (_Float16)val;
        }
        *(uint4*)&sEF[el][s0] = p.u;        // 16B-aligned b128 store
    }

    __syncthreads();

    // ---- Per-wave WMMA compute: wave w handles head h = w ----
    const int lane = tid & 31;
    const int h    = tid >> 5;            // 8 waves -> 8 heads
    const int half = lane >> 4;           // A/B half-wave group
    const int mn   = lane & 15;           // A: M row ; B/C: N col

    float part[4] = {0.f, 0.f, 0.f, 0.f};

    // Keep nt as a real loop: one iteration needs ~150 VGPRs, staying under
    // the 256-VGPR window (avoids s_set_vgpr_msb churn, allows 2+ waves/SIMD).
    #pragma unroll 1
    for (int nt = 0; nt < 4; ++nt) {      // g-tile: rows nt*16 .. nt*16+15
        const int row = nt * 16 + mn;

        // A fragments (16x32 f16): lane<16 -> K in {0..7,16..23}; lane>=16 -> +8.
        // Each fragment = two runs of 8 contiguous halves (16B each).
        v16h Aq0, Aq1, Ak0, Ak1, Ad;
        if constexpr (USE_WS) {
            const _Float16* wq = wq16 + h * N_FEAT * N_FEAT + row * N_FEAT + half * 8;
            const _Float16* wk = wk16 + h * N_FEAT * N_FEAT + row * N_FEAT + half * 8;
            const _Float16* wd = wd16 + h * N_FEAT * 32     + row * 32     + half * 8;
            Aq0 = load_frag16(wq);        // runs at +0 and +16 (K 0..7 / 16..23)
            Aq1 = load_frag16(wq + 32);   // ks=1: K 32..39 / 48..55 (+half*8)
            Ak0 = load_frag16(wk);
            Ak1 = load_frag16(wk + 32);
            Ad  = load_frag16(wd);
        } else {
            const float* Wqh = W_q  + h * N_FEAT * N_FEAT;
            const float* Wkh = W_k  + h * N_FEAT * N_FEAT;
            const float* Wdh = W_dk + h * N_FEAT * N_RBF_C;
            #pragma unroll
            for (int v = 0; v < 8; ++v) {
                int ka = half * 8 + ((v < 4) ? 2 * v : 2 * v + 8);
                float2 q0 = *(const float2*)(Wqh + row * N_FEAT + ka);
                float2 q1 = *(const float2*)(Wqh + row * N_FEAT + 32 + ka);
                float2 k0 = *(const float2*)(Wkh + row * N_FEAT + ka);
                float2 k1 = *(const float2*)(Wkh + row * N_FEAT + 32 + ka);
                Aq0[2 * v] = (_Float16)q0.x;  Aq0[2 * v + 1] = (_Float16)q0.y;
                Aq1[2 * v] = (_Float16)q1.x;  Aq1[2 * v + 1] = (_Float16)q1.y;
                Ak0[2 * v] = (_Float16)k0.x;  Ak0[2 * v + 1] = (_Float16)k0.y;
                Ak1[2 * v] = (_Float16)k1.x;  Ak1[2 * v + 1] = (_Float16)k1.y;
                float d0 = (ka     < N_RBF_C) ? Wdh[row * N_RBF_C + ka]     : 0.0f;
                float d1 = (ka + 1 < N_RBF_C) ? Wdh[row * N_RBF_C + ka + 1] : 0.0f;
                Ad[2 * v] = (_Float16)d0;     Ad[2 * v + 1] = (_Float16)d1;
            }
        }

        // Biases for the 8 g-rows of this lane's C fragment: two float4 loads each.
        float bq[8], bk[8], bd[8];
        {
            int g0 = h * N_FEAT + nt * 16 + half * 8;
            float4 q0 = *(const float4*)(b_q  + g0);
            float4 q1 = *(const float4*)(b_q  + g0 + 4);
            float4 k0 = *(const float4*)(b_k  + g0);
            float4 k1 = *(const float4*)(b_k  + g0 + 4);
            float4 d0 = *(const float4*)(b_dk + g0);
            float4 d1 = *(const float4*)(b_dk + g0 + 4);
            bq[0]=q0.x; bq[1]=q0.y; bq[2]=q0.z; bq[3]=q0.w;
            bq[4]=q1.x; bq[5]=q1.y; bq[6]=q1.z; bq[7]=q1.w;
            bk[0]=k0.x; bk[1]=k0.y; bk[2]=k0.z; bk[3]=k0.w;
            bk[4]=k1.x; bk[5]=k1.y; bk[6]=k1.z; bk[7]=k1.w;
            bd[0]=d0.x; bd[1]=d0.y; bd[2]=d0.z; bd[3]=d0.w;
            bd[4]=d1.x; bd[5]=d1.y; bd[6]=d1.z; bd[7]=d1.w;
        }

        #pragma unroll
        for (int st = 0; st < 4; ++st) {  // edge subtile: 16 edges
            const int rb = st * 16 + mn;  // LDS row = local edge for column N=mn

            // B fragments (32x16 f16): lane<16 -> K in {0..7,16..23}; lane>=16 -> +8.
            // Two runs of 8 halves per fragment -> 2 x ds_load_b128.
            const _Float16* pxi = &sXi[rb][half * 8];
            const _Float16* pxj = &sXj[rb][half * 8];
            const _Float16* pef = &sEF[rb][half * 8];
            v16h Bxi0 = load_frag16(pxi);
            v16h Bxi1 = load_frag16(pxi + 32);
            v16h Bxj0 = load_frag16(pxj);
            v16h Bxj1 = load_frag16(pxj + 32);
            v16h Bef  = load_frag16(pef);

            v8f accQ = {}, accK = {}, accD = {};
            accQ = __builtin_amdgcn_wmma_f32_16x16x32_f16(false, Aq0, false, Bxi0,
                                                          (short)0, accQ, false, false);
            accQ = __builtin_amdgcn_wmma_f32_16x16x32_f16(false, Aq1, false, Bxi1,
                                                          (short)0, accQ, false, false);
            accK = __builtin_amdgcn_wmma_f32_16x16x32_f16(false, Ak0, false, Bxj0,
                                                          (short)0, accK, false, false);
            accK = __builtin_amdgcn_wmma_f32_16x16x32_f16(false, Ak1, false, Bxj1,
                                                          (short)0, accK, false, false);
            accD = __builtin_amdgcn_wmma_f32_16x16x32_f16(false, Ad,  false, Bef,
                                                          (short)0, accD, false, false);

            // Fuse: p += (q+bq)*(k+bk)*silu(d+bd) over this fragment's 8 g-rows
            float p = 0.0f;
            #pragma unroll
            for (int i = 0; i < 8; ++i) {
                float q  = accQ[i] + bq[i];
                float kk = accK[i] + bk[i];
                float dv = accD[i] + bd[i];
                p += q * kk * silu_f(dv);
            }
            part[st] += p;
        }
    }

    // Combine the two half-wave g-row groups, apply outer silu, store.
    #pragma unroll
    for (int st = 0; st < 4; ++st) {
        float tot = part[st] + __shfl_xor(part[st], 16, 32);
        float w   = silu_f(tot);
        int e = ebase + st * 16 + mn;
        if (half == 0 && e < N_EDGES_C) {
            out[e * N_HEADS + h] = w;
        }
    }
}

extern "C" void kernel_launch(void* const* d_in, const int* in_sizes, int n_in,
                              void* d_out, int out_size, void* d_ws, size_t ws_size,
                              hipStream_t stream) {
    const float* dist = (const float*)d_in[0];
    const int*   nbrs = (const int*)  d_in[1];
    const float* x_i  = (const float*)d_in[2];
    const float* W_q  = (const float*)d_in[3];
    const float* b_q  = (const float*)d_in[4];
    const float* W_k  = (const float*)d_in[5];
    const float* b_k  = (const float*)d_in[6];
    const float* W_dk = (const float*)d_in[7];
    const float* b_dk = (const float*)d_in[8];
    float* out = (float*)d_out;

    const int blocks = (N_EDGES_C + E_TILE - 1) / E_TILE;   // 2344

    if (ws_size >= WS_NEED) {
        _Float16* wq16 = (_Float16*)d_ws;
        _Float16* wk16 = wq16 + WQ16_N;
        _Float16* wd16 = wk16 + WK16_N;
        convert_weights<<<dim3(WQ16_N / 256), dim3(256), 0, stream>>>(
            W_q, W_k, W_dk, wq16, wk16, wd16);
        painn_attn_wmma<true><<<dim3(blocks), dim3(256), 0, stream>>>(
            dist, nbrs, x_i, W_q, b_q, W_k, b_k, W_dk, b_dk,
            wq16, wk16, wd16, out);
    } else {
        painn_attn_wmma<false><<<dim3(blocks), dim3(256), 0, stream>>>(
            dist, nbrs, x_i, W_q, b_q, W_k, b_k, W_dk, b_dk,
            nullptr, nullptr, nullptr, out);
    }
}
